// Attention3d_8693013807287
// MI455X (gfx1250) — matrix operand order repaired and checked by reading, not compiled, not run on hardware
//
#include <hip/hip_runtime.h>
#include <hip/hip_bf16.h>

// ---------------------------------------------------------------------------
// Attention3d block for MI455X (gfx1250, wave32, WMMA f16 16x16x32 -> f32)
//
//   hn  = GroupNorm(x)                      (f32 -> f16)
//   h   = W_in  hn + b_in                   (WMMA GEMM, f16 out)
//   q/k/v = W_{q,k,v} h + b                 (WMMA GEMM, f16 out)
//   o   = flash-attention(q,k,v)            (WMMA, online softmax)
//   out = W_out o + b_out + x               (WMMA GEMM, f32 out)
//
// Fully unrolled softmax, scale folded into Q, double-buffered LDS staging,
// async global->LDS copies (GLOBAL_LOAD_ASYNC_TO_LDS_B128) for straight-copy
// tiles, s_wait_asynccnt before barriers.
// ---------------------------------------------------------------------------

typedef __attribute__((ext_vector_type(16))) _Float16 v16h;
typedef __attribute__((ext_vector_type(8)))  _Float16 v8h;
typedef __attribute__((ext_vector_type(4)))  _Float16 v4h;
typedef __attribute__((ext_vector_type(8)))  float    v8f;
typedef __attribute__((ext_vector_type(4)))  float    v4f;
typedef __attribute__((ext_vector_type(4)))  int      v4i;

#define CH     256
#define HEADS  4
#define HDIM   64
#define INNER  256
#define NTOK   4096
#define BATCH  2
#define NGROUP 32
#define EPSV   1e-5f

// ---------------------------------------------------------------------------
// Async global -> LDS (CDNA5 GLOBAL_LOAD_ASYNC_TO_LDS_B128), guarded.
// Builtin signature (from toolchain diagnostic): (v4i addrspace(1)*,
// v4i addrspace(3)*, imm int offset, imm int cpol). AS pointers are built via
// integer casts: low 32 bits of a generic LDS pointer are the LDS byte offset
// (flat aperture), generic global pointers are bit-identical to AS1.
// ---------------------------------------------------------------------------
#if defined(__has_builtin)
#if __has_builtin(__builtin_amdgcn_global_load_async_to_lds_b128)
#define ASYNC_LDS 1
#endif
#endif
#ifndef ASYNC_LDS
#define ASYNC_LDS 0
#endif

typedef __attribute__((address_space(1))) v4i* gas_v4i;
typedef __attribute__((address_space(3))) v4i* las_v4i;

__device__ inline void async_copy_b128(const _Float16* g, _Float16* l) {
#if ASYNC_LDS
  __builtin_amdgcn_global_load_async_to_lds_b128(
      (gas_v4i)(uintptr_t)g, (las_v4i)(uintptr_t)l, 0, 0);
#else
  *(v8h*)l = *(const v8h*)g;
#endif
}

__device__ inline void async_wait_all() {
#if ASYNC_LDS
  asm volatile("s_wait_asynccnt 0" ::: "memory");
#endif
}

// ---------------------------------------------------------------------------
// WMMA fragment helpers (layouts per CDNA5 ISA 7.12.2)
// ---------------------------------------------------------------------------
__device__ inline v16h make_v16(v8h lo, v8h hi) {
  return __builtin_shufflevector(lo, hi, 0, 1, 2, 3, 4, 5, 6, 7,
                                         8, 9, 10, 11, 12, 13, 14, 15);
}

// A fragment: 16x32 f16 (MxK) from row-major LDS [M][ld].
__device__ inline v16h load_a_frag(const _Float16* s, int ld, int mBase,
                                   int kBase, int lane) {
  const int m  = mBase + (lane & 15);
  const int kh = (lane >> 4) << 3;           // 0 or 8
  const _Float16* p = s + m * ld + kBase + kh;
  v8h lo = *(const v8h*)(p);                 // K = kh .. kh+7
  v8h hi = *(const v8h*)(p + 16);            // K = kh+16 .. kh+23
  return make_v16(lo, hi);
}

// B fragment: 32x16 f16 (KxN) from n-major (transposed) LDS Bt[N][ld].
__device__ inline v16h load_b_frag(const _Float16* s, int ld, int nBase,
                                   int kBase, int lane) {
  const int n  = nBase + (lane & 15);
  const int kb = kBase + ((lane >> 4) << 3); // 0 or 8
  const _Float16* p = s + n * ld + kb;
  v8h lo = *(const v8h*)(p);
  v8h hi = *(const v8h*)(p + 16);
  return make_v16(lo, hi);
}

__device__ inline v8f wmma_16x16x32(v16h a, v16h b, v8f c) {
  return __builtin_amdgcn_wmma_f32_16x16x32_f16(false, a, false, b,
                                                (short)0, c, false, false);
}

// ---------------------------------------------------------------------------
// Kernel 1: f32 -> f16 weight conversion (5 x 256x256 matrices)
// ---------------------------------------------------------------------------
__global__ __launch_bounds__(256) void cvt_kernel(const float* __restrict__ src,
                                                  _Float16* __restrict__ dst,
                                                  int n) {
  int i = (blockIdx.x * 256 + threadIdx.x) * 4;
  if (i < n) {
    v4f v = *(const v4f*)(src + i);
    v4h o;
#pragma unroll
    for (int j = 0; j < 4; ++j) o[j] = (_Float16)v[j];
    *(v4h*)(dst + i) = o;
  }
}

// ---------------------------------------------------------------------------
// Kernel 2: GroupNorm (B*NGROUP blocks, 256 threads). 8 ch x 4096 per group.
// ---------------------------------------------------------------------------
__global__ __launch_bounds__(256)
void gn_kernel(const float* __restrict__ x, const float* __restrict__ gw,
               const float* __restrict__ gb, _Float16* __restrict__ hn) {
  const int bg = blockIdx.x;
  const int b = bg / NGROUP, g = bg % NGROUP;
  const int cpg = CH / NGROUP;               // 8
  const int c0 = g * cpg;
  const size_t base = ((size_t)b * CH + c0) * NTOK;
  const int t = threadIdx.x;
  const int TOT = cpg * NTOK;                // 32768

  float s = 0.f, ss = 0.f;
  for (int i = t * 4; i < TOT; i += 256 * 4) {
    v4f v = *(const v4f*)(x + base + i);
#pragma unroll
    for (int j = 0; j < 4; ++j) { s += v[j]; ss += v[j] * v[j]; }
  }
#pragma unroll
  for (int m = 1; m < 32; m <<= 1) {
    s  += __shfl_xor(s,  m, 32);
    ss += __shfl_xor(ss, m, 32);
  }
  __shared__ float rs[8], rss[8];
  const int wave = t >> 5, lane = t & 31;
  if (lane == 0) { rs[wave] = s; rss[wave] = ss; }
  __syncthreads();
  if (t == 0) {
    float a = 0.f, c = 0.f;
#pragma unroll
    for (int i = 0; i < 8; ++i) { a += rs[i]; c += rss[i]; }
    rs[0] = a; rss[0] = c;
  }
  __syncthreads();
  const float mean = rs[0] / (float)TOT;
  const float var  = rss[0] / (float)TOT - mean * mean;
  const float inv  = rsqrtf(var + EPSV);

  for (int i = t * 4; i < TOT; i += 256 * 4) {
    const int c = c0 + i / NTOK;             // i..i+3 stay within one channel
    const float a = gw[c] * inv;
    const float bb = gb[c] - mean * a;
    v4f v = *(const v4f*)(x + base + i);
    v4h o;
#pragma unroll
    for (int j = 0; j < 4; ++j) o[j] = (_Float16)(v[j] * a + bb);
    *(v4h*)(hn + base + i) = o;
  }
}

// ---------------------------------------------------------------------------
// Kernel 3: WMMA GEMM  Y[b,o,n] = sum_c W[o,c] X[b,c,n] + bias[o]  (+resid)
// M = K = 256, N = 4096/batch. 128x128 tile, 8 waves of 64x32.
// Double-buffered LDS; A tile staged with async global->LDS copies.
// ---------------------------------------------------------------------------
template <bool FINAL>
__global__ __launch_bounds__(256)
void gemm256_kernel(const _Float16* __restrict__ W, const float* __restrict__ bias,
                    const _Float16* __restrict__ X, _Float16* __restrict__ Yh,
                    float* __restrict__ Yf, const float* __restrict__ resid) {
  const int b  = blockIdx.z;
  const int m0 = blockIdx.y * 128;
  const int n0 = blockIdx.x * 128;
  const _Float16* Xb = X + (size_t)b * CH * NTOK;

  __shared__ _Float16 As[2][128][32];        // W tile, row-major [m][k]
  __shared__ _Float16 Bt[2][128][32];        // X tile, transposed [n][k]

  const int t = threadIdx.x, lane = t & 31, wave = t >> 5;
  const int wm = (wave >> 2) * 64;           // 0 or 64
  const int wn = (wave & 3) * 32;            // 0..96

  // staging geometry (fixed per thread)
  const int am = t >> 1, akb = (t & 1) * 16; // A: 16 halves of one W row
  const int bk = t >> 3, bnb = (t & 7) * 16; // B: 16 cols at one k row

  v8h breg0, breg1;

  auto stage_A = [&](int buf, int k0) {
    const _Float16* src = W + (size_t)(m0 + am) * CH + k0 + akb;
    async_copy_b128(src,     &As[buf][am][akb]);
    async_copy_b128(src + 8, &As[buf][am][akb + 8]);
  };
  auto load_B = [&](int k0) {
    const _Float16* src = Xb + (size_t)(k0 + bk) * NTOK + n0 + bnb;
    breg0 = *(const v8h*)(src);
    breg1 = *(const v8h*)(src + 8);
  };
  auto store_B = [&](int buf) {
#pragma unroll
    for (int j = 0; j < 8; ++j) Bt[buf][bnb + j][bk]     = breg0[j];
#pragma unroll
    for (int j = 0; j < 8; ++j) Bt[buf][bnb + 8 + j][bk] = breg1[j];
  };

  v8f acc[4][2] = {};

  stage_A(0, 0);
  load_B(0);
  store_B(0);
  async_wait_all();
  __syncthreads();

  for (int it = 0; it < CH / 32; ++it) {
    const int buf = it & 1;
    if (it + 1 < CH / 32) {                  // kick off next tile early
      stage_A(buf ^ 1, (it + 1) * 32);
      load_B((it + 1) * 32);
    }

    v16h bf[2];
#pragma unroll
    for (int j = 0; j < 2; ++j)
      bf[j] = load_b_frag(&Bt[buf][0][0], 32, wn + j * 16, 0, lane);
#pragma unroll
    for (int i = 0; i < 4; ++i) {
      v16h af = load_a_frag(&As[buf][0][0], 32, wm + i * 16, 0, lane);
#pragma unroll
      for (int j = 0; j < 2; ++j)
        acc[i][j] = wmma_16x16x32(af, bf[j], acc[i][j]);
    }

    if (it + 1 < CH / 32) store_B(buf ^ 1);
    async_wait_all();
    __syncthreads();
  }

  // Epilogue: C/D layout -> lane holds n = lane&15, rows v (+8 upper half)
  const int mup = (lane >> 4) << 3;
  const int nc  = lane & 15;
#pragma unroll
  for (int i = 0; i < 4; ++i)
#pragma unroll
    for (int j = 0; j < 2; ++j)
#pragma unroll
      for (int vv = 0; vv < 8; ++vv) {
        const int o = m0 + wm + i * 16 + vv + mup;
        const int n = n0 + wn + j * 16 + nc;
        const size_t idx = ((size_t)b * CH + o) * NTOK + n;
        const float val = acc[i][j][vv] + bias[o];
        if (FINAL) Yf[idx] = val + resid[idx];
        else       Yh[idx] = (_Float16)val;
      }
}

// ---------------------------------------------------------------------------
// Kernel 4: flash attention.  grid (NTOK/64, BATCH*HEADS), 128 threads.
// 64 queries/block, streamed 64-key blocks, online softmax, double-buffered
// K/V staging (V via async global->LDS). Scale folded into Q (exact, 2^-3).
// ---------------------------------------------------------------------------
__global__ __launch_bounds__(128)
void flash_kernel(const _Float16* __restrict__ Q, const _Float16* __restrict__ K,
                  const _Float16* __restrict__ V, _Float16* __restrict__ O) {
  const int bh = blockIdx.y;
  const size_t base = (size_t)bh * HDIM * NTOK;   // (b*INNER + h*64)*NTOK
  const int i0 = blockIdx.x * 64;
  const int t = threadIdx.x, lane = t & 31, w = t >> 5;   // 4 waves x 16 rows

  __shared__ _Float16 Qs[64][72];            // [i][d] (transposed, pre-scaled)
  __shared__ _Float16 Kt[2][64][72];         // [j][d] (transposed)
  __shared__ _Float16 Vs[2][64][72];         // [d][j] (natural)
  __shared__ _Float16 Ps[64][72];            // [i][j] softmax weights

  const int sd = t >> 1, sjb = (t & 1) * 32; // staging geometry
  v8h kreg[4];

  auto load_K = [&](int j0) {
    const _Float16* ks = K + base + (size_t)sd * NTOK + j0 + sjb;
#pragma unroll
    for (int c = 0; c < 4; ++c) kreg[c] = *(const v8h*)(ks + c * 8);
  };
  auto store_K = [&](int buf) {
#pragma unroll
    for (int c = 0; c < 4; ++c)
#pragma unroll
      for (int j = 0; j < 8; ++j) Kt[buf][sjb + c * 8 + j][sd] = kreg[c][j];
  };
  auto stage_V = [&](int buf, int j0) {
    const _Float16* vs = V + base + (size_t)sd * NTOK + j0 + sjb;
#pragma unroll
    for (int c = 0; c < 4; ++c)
      async_copy_b128(vs + c * 8, &Vs[buf][sd][sjb + c * 8]);
  };

  {   // stage Q transposed, folding in the 1/sqrt(HDIM)=0.125 scale (exact)
    const _Float16 qsc = (_Float16)0.125f;
    const _Float16* src = Q + base + (size_t)sd * NTOK + i0 + sjb;
#pragma unroll
    for (int c = 0; c < 4; ++c) {
      v8h x = *(const v8h*)(src + c * 8);
#pragma unroll
      for (int j = 0; j < 8; ++j) Qs[sjb + c * 8 + j][sd] = x[j] * qsc;
    }
  }

  v8f oacc[4] = {};
  float mrow[8], lrow[8];
#pragma unroll
  for (int r = 0; r < 8; ++r) { mrow[r] = -1e30f; lrow[r] = 0.f; }

  load_K(0);
  store_K(0);
  stage_V(0, 0);
  async_wait_all();
  __syncthreads();

  for (int it = 0; it < NTOK / 64; ++it) {
    const int buf = it & 1;
    if (it + 1 < NTOK / 64) {                // start staging next key block
      load_K((it + 1) * 64);
      stage_V(buf ^ 1, (it + 1) * 64);
    }

    // S = (Q*scale) K^T : wave w owns rows [w*16, w*16+16), all 64 cols
    v8f s[4] = {};
#pragma unroll
    for (int kd = 0; kd < 64; kd += 32) {
      v16h aq = load_a_frag(&Qs[0][0], 72, w * 16, kd, lane);
#pragma unroll
      for (int ns = 0; ns < 4; ++ns) {
        v16h bk = load_b_frag(&Kt[buf][0][0], 72, ns * 16, kd, lane);
        s[ns] = wmma_16x16x32(aq, bk, s[ns]);
      }
    }

    // Online softmax: per-row max/sum via 16-lane xor-shuffle reductions
    // (masks < 16 keep the two 16-lane halves of the D layout separate).
#pragma unroll
    for (int vv = 0; vv < 8; ++vv) {
      float rm = -1e30f;
#pragma unroll
      for (int ns = 0; ns < 4; ++ns) rm = fmaxf(rm, s[ns][vv]);
#pragma unroll
      for (int msk = 1; msk < 16; msk <<= 1)
        rm = fmaxf(rm, __shfl_xor(rm, msk, 32));
      const float newm = fmaxf(mrow[vv], rm);
      const float corr = __expf(mrow[vv] - newm);
      lrow[vv] *= corr;
#pragma unroll
      for (int ns = 0; ns < 4; ++ns) oacc[ns][vv] *= corr;
      float rsum = 0.f;
#pragma unroll
      for (int ns = 0; ns < 4; ++ns) {
        const float p = __expf(s[ns][vv] - newm);
        s[ns][vv] = p;
        rsum += p;
      }
#pragma unroll
      for (int msk = 1; msk < 16; msk <<= 1)
        rsum += __shfl_xor(rsum, msk, 32);
      lrow[vv] += rsum;
      mrow[vv] = newm;
    }

    if (it + 1 < NTOK / 64) store_K(buf ^ 1);

    {   // spill P (f16) to LDS; rows are wave-private, DS ops are in-order
      const int ir = w * 16 + ((lane >> 4) << 3);
      const int jc = lane & 15;
#pragma unroll
      for (int ns = 0; ns < 4; ++ns)
#pragma unroll
        for (int vv = 0; vv < 8; ++vv)
          Ps[ir + vv][ns * 16 + jc] = (_Float16)s[ns][vv];
    }

    // O += P V^T : B-matrix (k=j, n=d) reads Vs[d][j] directly
#pragma unroll
    for (int kj = 0; kj < 64; kj += 32) {
      v16h ap = load_a_frag(&Ps[0][0], 72, w * 16, kj, lane);
#pragma unroll
      for (int ds = 0; ds < 4; ++ds) {
        v16h bv = load_b_frag(&Vs[buf][0][0], 72, ds * 16, kj, lane);
        oacc[ds] = wmma_16x16x32(ap, bv, oacc[ds]);
      }
    }

    async_wait_all();
    __syncthreads();                         // next iter reads buf^1
  }

  {   // normalize and store transposed: O[d][i]
    const int jc  = lane & 15;
    const int mup = (lane >> 4) << 3;
#pragma unroll
    for (int ds = 0; ds < 4; ++ds)
#pragma unroll
      for (int vv = 0; vv < 8; ++vv) {
        const int d  = ds * 16 + jc;
        const int ii = i0 + w * 16 + vv + mup;
        O[base + (size_t)d * NTOK + ii] = (_Float16)(oacc[ds][vv] / lrow[vv]);
      }
  }
}

// ---------------------------------------------------------------------------
// Host launcher
// ---------------------------------------------------------------------------
extern "C" void kernel_launch(void* const* d_in, const int* in_sizes, int n_in,
                              void* d_out, int out_size, void* d_ws, size_t ws_size,
                              hipStream_t stream) {
  const float* x     = (const float*)d_in[0];
  const float* gn_w  = (const float*)d_in[1];
  const float* gn_b  = (const float*)d_in[2];
  const float* w_in  = (const float*)d_in[3];
  const float* b_in  = (const float*)d_in[4];
  const float* w_q   = (const float*)d_in[5];
  const float* b_q   = (const float*)d_in[6];
  const float* w_k   = (const float*)d_in[7];
  const float* b_k   = (const float*)d_in[8];
  const float* w_v   = (const float*)d_in[9];
  const float* b_v   = (const float*)d_in[10];
  const float* w_out = (const float*)d_in[11];
  const float* b_out = (const float*)d_in[12];

  char* ws = (char*)d_ws;
  const size_t WSZ = (size_t)INNER * CH * sizeof(_Float16);        // 128 KB
  const size_t ACT = (size_t)BATCH * CH * NTOK * sizeof(_Float16); // 4 MB
  _Float16* w_in16  = (_Float16*)(ws + 0 * WSZ);
  _Float16* w_q16   = (_Float16*)(ws + 1 * WSZ);
  _Float16* w_k16   = (_Float16*)(ws + 2 * WSZ);
  _Float16* w_v16   = (_Float16*)(ws + 3 * WSZ);
  _Float16* w_out16 = (_Float16*)(ws + 4 * WSZ);
  size_t off = 8 * WSZ;                                            // 1 MB align
  _Float16* hn16 = (_Float16*)(ws + off); off += ACT;
  _Float16* h16  = (_Float16*)(ws + off); off += ACT;
  _Float16* q16  = (_Float16*)(ws + off); off += ACT;
  _Float16* k16  = (_Float16*)(ws + off); off += ACT;
  _Float16* v16  = (_Float16*)(ws + off); off += ACT;
  _Float16* o16  = (_Float16*)(ws + off); off += ACT;

  const int WN = INNER * CH;                 // 65536
  cvt_kernel<<<64, 256, 0, stream>>>(w_in,  w_in16,  WN);
  cvt_kernel<<<64, 256, 0, stream>>>(w_q,   w_q16,   WN);
  cvt_kernel<<<64, 256, 0, stream>>>(w_k,   w_k16,   WN);
  cvt_kernel<<<64, 256, 0, stream>>>(w_v,   w_v16,   WN);
  cvt_kernel<<<64, 256, 0, stream>>>(w_out, w_out16, WN);

  gn_kernel<<<BATCH * NGROUP, 256, 0, stream>>>(x, gn_w, gn_b, hn16);

  dim3 ggrid(NTOK / 128, CH / 128, BATCH);   // (32, 2, 2)
  gemm256_kernel<false><<<ggrid, 256, 0, stream>>>(w_in16, b_in, hn16, h16, nullptr, nullptr);
  gemm256_kernel<false><<<ggrid, 256, 0, stream>>>(w_q16,  b_q,  h16,  q16, nullptr, nullptr);
  gemm256_kernel<false><<<ggrid, 256, 0, stream>>>(w_k16,  b_k,  h16,  k16, nullptr, nullptr);
  gemm256_kernel<false><<<ggrid, 256, 0, stream>>>(w_v16,  b_v,  h16,  v16, nullptr, nullptr);

  flash_kernel<<<dim3(NTOK / 64, BATCH * HEADS), 128, 0, stream>>>(q16, k16, v16, o16);

  gemm256_kernel<true><<<ggrid, 256, 0, stream>>>(w_out16, b_out, o16, nullptr,
                                                  (float*)d_out, x);
}
